// ResInvProjector_24730421690454
// MI455X (gfx1250) — matrix operand order repaired and checked by reading, not compiled, not run on hardware
//
#include <hip/hip_runtime.h>

typedef __attribute__((ext_vector_type(16))) _Float16 v16h;
typedef __attribute__((ext_vector_type(8)))  _Float16 v8h;
typedef __attribute__((ext_vector_type(8)))  float    v8f;

#define NPTS    (8 * 4096)        // 32768 points
#define NCHUNK  128               // K-chunks of 32 (virtual K = 64*64)
#define NK      64                // k index = chunk/2
#define NKB     65                // +1 bias pair (b2 folded in, h == 1)
#define CHUNK_HALVES 2048         // 4 tiles * 32 lanes * 16 halves
#define STAGE_HALVES (2 * CHUNK_HALVES)   // one k = 2 chunks = 8KB
#define BPACK_HALVES ((NCHUNK + 2) * CHUNK_HALVES)

// CDNA5 async global->LDS path (ASYNCcnt) if this toolchain exposes it.
#if defined(__gfx1250__) && __has_builtin(__builtin_amdgcn_global_load_async_to_lds_b128)
#define USE_ASYNC_LDS 1
#else
#define USE_ASYNC_LDS 0
#endif

#if USE_ASYNC_LDS
// Probe-derived signature: (int4 AS1* src, int4 AS3* dst, imm offset, imm cpol)
typedef int b128_t __attribute__((vector_size(16)));
__device__ __forceinline__ void async_copy16(const void* gsrc, void* ldst) {
    __builtin_amdgcn_global_load_async_to_lds_b128(
        (__attribute__((address_space(1))) b128_t*)gsrc,
        (__attribute__((address_space(3))) b128_t*)ldst, 0, 0);
}
#endif

__device__ __forceinline__ void wait_async_all() {
#if defined(__gfx1250__)
#if __has_builtin(__builtin_amdgcn_s_wait_asynccnt)
    __builtin_amdgcn_s_wait_asynccnt(0);
#else
    asm volatile("s_wait_asynccnt 0" ::: "memory");
#endif
#endif
}

// ---------------------------------------------------------------------------
// Kernel 0: pack B[kk,o] = W2[k, i*64+o] (kk = k*64+i) + b2 as 2 trailing
// chunks, in WMMA B-operand order: [chunk][tile][lane(=N col)][16 K halves].
// ---------------------------------------------------------------------------
__global__ __launch_bounds__(256) void pack_B_kernel(
    const float* __restrict__ W2,   // [64, 4096]
    const float* __restrict__ b2,   // [4096]
    _Float16* __restrict__ Bp)
{
    int idx = blockIdx.x * 256 + threadIdx.x;
    if (idx >= BPACK_HALVES) return;
    int j    = idx & 15;           // element: K row (table order)
    int lane = (idx >> 4) & 31;    // lane: N = lane&15, half = lane>>4
    int t    = (idx >> 9) & 3;     // N tile
    int c    = idx >> 11;          // K chunk
    int krow = (j < 8) ? ((lane >> 4) * 8 + j) : (16 + (lane >> 4) * 8 + (j - 8));
    int o    = t * 16 + (lane & 15);
    float v;
    if (c < NCHUNK) {
        int kk = c * 32 + krow;    // kk = k*64 + i
        int k  = kk >> 6;
        int i  = kk & 63;
        v = W2[k * 4096 + i * 64 + o];
    } else {
        int i = (c - NCHUNK) * 32 + krow;
        v = b2[i * 64 + o];
    }
    Bp[idx] = (_Float16)v;
}

// ---------------------------------------------------------------------------
// Kernel 1: WG = 256 threads = 8 waves; wave owns 2 M-tiles (32 points),
// WG covers 256 points. B double-buffered through LDS (async copy when
// available), one k (2 chunks = 8KB) per stage, 16 WMMAs per iteration.
// ---------------------------------------------------------------------------
__global__ __launch_bounds__(256) void projector_wmma_kernel(
    const float* __restrict__ x,     // [NPTS, 64]
    const float* __restrict__ grid,  // [NPTS, 3]
    const float* __restrict__ W1,    // [3, 64]
    const float* __restrict__ b1,    // [64]
    const _Float16* __restrict__ Bp, // packed B
    float* __restrict__ out)         // [NPTS, 64]
{
    __shared__ _Float16 h_sh[256 * 64];    // 32KB: gelu(grid@W1+b1) as f16
    __shared__ _Float16 xb_sh[256 * 64];   // 32KB: x (preload), then B stage

    const int tid     = threadIdx.x;
    const int wg_base = blockIdx.x * 256;

    // ---- Phase 1: one point per thread: h = gelu(grid@W1+b1), x -> f16
    {
        const int   gpt = wg_base + tid;
        const float g0  = grid[gpt * 3 + 0];
        const float g1  = grid[gpt * 3 + 1];
        const float g2  = grid[gpt * 3 + 2];
        #pragma unroll 4
        for (int k = 0; k < 64; ++k) {
            float pre = g0 * W1[k] + g1 * W1[64 + k] + g2 * W1[128 + k] + b1[k];
            float ge  = 0.5f * pre * (1.0f + erff(pre * 0.70710678118654752f));
            h_sh[tid * 64 + k] = (_Float16)ge;
        }
        const float4* xv = (const float4*)(x + (size_t)gpt * 64);
        #pragma unroll
        for (int q = 0; q < 16; ++q) {
            float4 v = xv[q];
            int b = tid * 64 + q * 4;
            xb_sh[b + 0] = (_Float16)v.x;
            xb_sh[b + 1] = (_Float16)v.y;
            xb_sh[b + 2] = (_Float16)v.z;
            xb_sh[b + 3] = (_Float16)v.w;
        }
    }
    __syncthreads();

    // ---- Per-lane A-operand x slices (16-bit A layout: half j<8 -> K=g*8+j,
    //      half j>=8 -> K=16+g*8+(j-8); xlo: i-base 0, xhi: i-base 32).
    const int lane = tid & 31;
    const int wave = tid >> 5;
    const int g    = lane >> 4;
    const int m    = lane & 15;
    const int pA   = wave * 32 + m;       // first M-tile's point (local)
    const int pB   = pA + 16;             // second M-tile's point (local)

    union V16 { v16h v; v8h h[2]; };
    V16 xloA, xhiA, xloB, xhiB;
    {
        const _Float16* xr = &xb_sh[pA * 64];
        xloA.h[0] = *(const v8h*)(xr + 0  + g * 8);
        xloA.h[1] = *(const v8h*)(xr + 16 + g * 8);
        xhiA.h[0] = *(const v8h*)(xr + 32 + g * 8);
        xhiA.h[1] = *(const v8h*)(xr + 48 + g * 8);
        xr = &xb_sh[pB * 64];
        xloB.h[0] = *(const v8h*)(xr + 0  + g * 8);
        xloB.h[1] = *(const v8h*)(xr + 16 + g * 8);
        xhiB.h[0] = *(const v8h*)(xr + 32 + g * 8);
        xhiB.h[1] = *(const v8h*)(xr + 48 + g * 8);
    }
    __syncthreads();   // all x reads done -> xb_sh becomes the B stage

    _Float16*   stage = xb_sh;            // 2 buffers x STAGE_HALVES
    const v16h* Bv    = (const v16h*)Bp;  // 32B granules of packed B

    // Prologue: stage k=0 (each thread copies one 32B granule)
#if USE_ASYNC_LDS
    async_copy16(Bv + tid,                   &stage[tid * 16]);
    async_copy16((const v8h*)(Bv + tid) + 1, &stage[tid * 16 + 8]);
    wait_async_all();
#else
    *(v16h*)&stage[tid * 16] = Bv[tid];
#endif
    __syncthreads();

    v8f aA0 = {}, aA1 = {}, aA2 = {}, aA3 = {};
    v8f aB0 = {}, aB1 = {}, aB2 = {}, aB3 = {};

    for (int k = 0; k < NKB; ++k) {
        const int buf = k & 1;
        const bool have_nxt = (k + 1) < NKB;

        // Kick off the next stage's copy immediately; it completes in the
        // background while this iteration's 16 WMMAs run.
#if USE_ASYNC_LDS
        if (have_nxt) {
            const v16h* src = Bv + (size_t)(k + 1) * 256 + tid;
            _Float16*   dst = &stage[(buf ^ 1) * STAGE_HALVES + tid * 16];
            async_copy16(src, dst);
            async_copy16((const v8h*)src + 1, dst + 8);
        }
#else
        v16h nxt;
        if (have_nxt) nxt = Bv[(size_t)(k + 1) * 256 + tid];
#endif
        if (k + 2 < NKB)
            __builtin_prefetch((const void*)(Bv + (size_t)(k + 2) * 256 + tid), 0, 0);

        const _Float16 hA = (k < NK) ? h_sh[pA * 64 + k] : (_Float16)1.0f;
        const _Float16 hB = (k < NK) ? h_sh[pB * 64 + k] : (_Float16)1.0f;

        const _Float16* sb = &stage[buf * STAGE_HALVES + lane * 16];
        #pragma unroll
        for (int ch = 0; ch < 2; ++ch) {          // chunk = 2k + ch
            const _Float16* cb = sb + ch * CHUNK_HALVES;
            v16h b0  = *(const v16h*)(cb + 0 * 512);
            v16h b1v = *(const v16h*)(cb + 1 * 512);
            v16h b2v = *(const v16h*)(cb + 2 * 512);
            v16h b3v = *(const v16h*)(cb + 3 * 512);

            v16h aAv = (ch ? xhiA.v : xloA.v) * hA;   // v_pk_mul_f16 splat
            v16h aBv = (ch ? xhiB.v : xloB.v) * hB;

            aA0 = __builtin_amdgcn_wmma_f32_16x16x32_f16(false, aAv, false, b0,  (short)0, aA0, false, false);
            aA1 = __builtin_amdgcn_wmma_f32_16x16x32_f16(false, aAv, false, b1v, (short)0, aA1, false, false);
            aA2 = __builtin_amdgcn_wmma_f32_16x16x32_f16(false, aAv, false, b2v, (short)0, aA2, false, false);
            aA3 = __builtin_amdgcn_wmma_f32_16x16x32_f16(false, aAv, false, b3v, (short)0, aA3, false, false);
            aB0 = __builtin_amdgcn_wmma_f32_16x16x32_f16(false, aBv, false, b0,  (short)0, aB0, false, false);
            aB1 = __builtin_amdgcn_wmma_f32_16x16x32_f16(false, aBv, false, b1v, (short)0, aB1, false, false);
            aB2 = __builtin_amdgcn_wmma_f32_16x16x32_f16(false, aBv, false, b2v, (short)0, aB2, false, false);
            aB3 = __builtin_amdgcn_wmma_f32_16x16x32_f16(false, aBv, false, b3v, (short)0, aB3, false, false);
        }

#if USE_ASYNC_LDS
        wait_async_all();          // our granules landed in LDS
#else
        if (have_nxt)
            *(v16h*)&stage[(buf ^ 1) * STAGE_HALVES + tid * 16] = nxt;
#endif
        __syncthreads();           // everyone's granules visible; cur reusable
    }

    // ---- Store per C/D layout: VGPR r -> M = g*8+r, lane%16 -> N col
    #pragma unroll
    for (int r = 0; r < 8; ++r) {
        const int rowA = wg_base + wave * 32 + g * 8 + r;
        const int rowB = rowA + 16;
        float* oA = out + (size_t)rowA * 64 + m;
        float* oB = out + (size_t)rowB * 64 + m;
        oA[0]  = aA0[r];  oA[16] = aA1[r];  oA[32] = aA2[r];  oA[48] = aA3[r];
        oB[0]  = aB0[r];  oB[16] = aB1[r];  oB[32] = aB2[r];  oB[48] = aB3[r];
    }
}

// ---------------------------------------------------------------------------
extern "C" void kernel_launch(void* const* d_in, const int* in_sizes, int n_in,
                              void* d_out, int out_size, void* d_ws, size_t ws_size,
                              hipStream_t stream) {
    const float* x    = (const float*)d_in[0];
    const float* grid = (const float*)d_in[1];
    const float* W1   = (const float*)d_in[2];
    const float* b1   = (const float*)d_in[3];
    const float* W2   = (const float*)d_in[4];
    const float* b2   = (const float*)d_in[5];
    float* out        = (float*)d_out;
    _Float16* Bp      = (_Float16*)d_ws;   // 532,480 bytes

    pack_B_kernel<<<(BPACK_HALVES + 255) / 256, 256, 0, stream>>>(W2, b2, Bp);
    projector_wmma_kernel<<<NPTS / 256, 256, 0, stream>>>(x, grid, W1, b1, Bp, out);
}
